// EncoderLayer_60945585930290
// MI455X (gfx1250) — compile-verified, matrix operand order checked by reading, not run on hardware
//
#include <hip/hip_runtime.h>
#include <cfloat>
#include <cmath>
#include <cstdint>

// Problem constants (from reference)
#define B_    4
#define L_    2048
#define D_    256
#define H_    8
#define DH_   32
#define SK_   40      // SAMPLE_K
#define NT_   40      // N_TOP
#define EPS_  1e-5f
#define ROWS_ (B_ * L_)   // 8192

typedef __bf16 bf16;
typedef __attribute__((ext_vector_type(8)))  bf16  v8bf;
typedef __attribute__((ext_vector_type(16))) bf16  v16bf;
typedef __attribute__((ext_vector_type(8)))  float v8f;

union Frag16 { v16bf v; v8bf h[2]; };

// ---------------------------------------------------------------------------
// CDNA5 async global->LDS (ASYNCcnt-tracked), per CDNA5 ISA
// and §15.18.3 (GLOBAL_LOAD_ASYNC_TO_LDS_B128, opcode 98).  INST_OFFSET is
// applied to BOTH the LDS and global addresses, so the +16B chunk needs no
// extra address math.  Generic pointers to LDS carry the DS offset in their
// low 32 bits on AMDGPU.
// ---------------------------------------------------------------------------
__device__ __forceinline__ unsigned lds_off(const void* p) {
  return (unsigned)(uintptr_t)p;
}
template <int OFF>
__device__ __forceinline__ void async_load_b128(unsigned dst_lds, const void* gsrc) {
  unsigned long long ga = (unsigned long long)(uintptr_t)gsrc;
  asm volatile("global_load_async_to_lds_b128 %0, %1, off offset:%2"
               :: "v"(dst_lds), "v"(ga), "i"(OFF)
               : "memory");
}
template <int N>
__device__ __forceinline__ void s_wait_async() {
  asm volatile("s_wait_asynccnt %0" :: "i"(N) : "memory");
}

// ---------------------------------------------------------------------------
// Precision converts.  Weights are pre-transposed to N-major bf16 so the GEMM
// stages BOTH operands with vectorized 16B async copies (no scalar ds stores).
// ---------------------------------------------------------------------------
__global__ void f32_to_bf16_kernel(const float* __restrict__ in,
                                   bf16* __restrict__ out, int n) {
  int i = blockIdx.x * blockDim.x + threadIdx.x;
  if (i < n) out[i] = (bf16)in[i];
}

// in: (R,C) row-major fp32 -> out: (C,R) row-major bf16
__global__ void transpose_to_bf16_kernel(const float* __restrict__ in,
                                         bf16* __restrict__ out, int R, int C) {
  int i = blockIdx.x * blockDim.x + threadIdx.x;
  if (i < R * C) {
    int r = i / C, c = i % C;
    out[(size_t)c * R + r] = (bf16)in[i];
  }
}

// ---------------------------------------------------------------------------
// bf16 WMMA GEMM:  C(MxN,f32) = A(MxK,bf16) * B^T + bias, B given N-major (NxK)
// 256 threads = 8 waves; block tile 128x64; wave tile 32x32 (2x2 WMMA tiles).
// Double-buffered LDS, tiles staged with global_load_async_to_lds_b128 and
// pipelined on ASYNCcnt (branch-free steady state, last tile peeled).
// Optional ReLU; optional bf16 output stream.
// ---------------------------------------------------------------------------
__global__ void __launch_bounds__(256) gemm_bf16_wmma_kernel(
    const bf16* __restrict__ A, const bf16* __restrict__ Bm,
    const float* __restrict__ bias,
    float* __restrict__ Cf, bf16* __restrict__ Cb,
    int M, int N, int K, int relu)
{
  // 32 K + pad 8 -> 80B rows keep every fragment sub-load 16B aligned
  __shared__ bf16 As[2][128][40];
  __shared__ bf16 Bs[2][64][40];    // N-major: Bs[buf][n][k]

  const int tid  = threadIdx.x;
  const int wave = tid >> 5;
  const int lane = tid & 31;
  const int half = lane >> 4;       // 0/1 per WMMA 16-bit layout
  const int l16  = lane & 15;

  const int row0 = blockIdx.y * 128;
  const int col0 = blockIdx.x * 64;
  const int wm   = (wave & 3) * 32; // wave row offset inside block tile
  const int wn   = (wave >> 2) * 32;// wave col offset inside block tile

  // Staging assignments: A tile 128x32 -> 2 x 16B per thread,
  //                      B tile 64x32  -> 1 x 16B per thread.
  const int ar = tid >> 1;
  const int ac = (tid & 1) * 16;
  const int bn = tid >> 2;
  const int bc = (tid & 3) * 8;
  const bf16* aCur = A  + (size_t)(row0 + ar) * K + ac;
  const bf16* bCur = Bm + (size_t)(col0 + bn) * K + bc;

  const unsigned laA[2] = { lds_off(&As[0][ar][ac]), lds_off(&As[1][ar][ac]) };
  const unsigned laB[2] = { lds_off(&Bs[0][bn][bc]), lds_off(&Bs[1][bn][bc]) };

  const int nTiles = K >> 5;

  // issue one K-tile of async copies (3 ASYNCcnt ops per wave), advance ptrs
  auto issue_tile = [&](int buf) {
    async_load_b128<0>(laA[buf], aCur);
    async_load_b128<16>(laA[buf], aCur);   // LDS+16 and global+16
    async_load_b128<0>(laB[buf], bCur);
    aCur += 32;
    bCur += 32;
  };

  v8f acc[2][2] = {};

  auto compute_tile = [&](int buf) {
    // Fragment loads per ISA 16-bit layout:
    //  lane<16: K {0..7,16..23}; lane>=16: K {8..15,24..31}
    Frag16 a[2], b[2];
#pragma unroll
    for (int i = 0; i < 2; ++i) {
      const bf16* p = &As[buf][wm + i * 16 + l16][half * 8];
      a[i].h[0] = *(const v8bf*)p;
      a[i].h[1] = *(const v8bf*)(p + 16);
    }
#pragma unroll
    for (int j = 0; j < 2; ++j) {
      const bf16* p = &Bs[buf][wn + j * 16 + l16][half * 8];
      b[j].h[0] = *(const v8bf*)p;
      b[j].h[1] = *(const v8bf*)(p + 16);
    }
#pragma unroll
    for (int i = 0; i < 2; ++i)
#pragma unroll
      for (int j = 0; j < 2; ++j)
        acc[i][j] = __builtin_amdgcn_wmma_f32_16x16x32_bf16(
            false, a[i].v, false, b[j].v, (short)0, acc[i][j], false, false);
  };

  issue_tile(0);
  int t = 0;
  for (; t < nTiles - 1; ++t) {          // branch-free steady state
    issue_tile((t + 1) & 1);
    s_wait_async<3>();   // tile t's 3 ops retired (async loads retire in order)
    __syncthreads();
    compute_tile(t & 1);
    __syncthreads();     // protect buffer reuse by the next issue
  }
  s_wait_async<0>();                     // peeled last tile
  __syncthreads();
  compute_tile(t & 1);

  // Epilogue: bias (+ReLU) + fp32/bf16 stores.  C layout: VGPR e ->
  // row = half*8+e, col = lane&15 within each 16x16 tile.
#pragma unroll
  for (int i = 0; i < 2; ++i) {
#pragma unroll
    for (int j = 0; j < 2; ++j) {
      int colg = col0 + wn + j * 16 + l16;
      float bv = bias ? bias[colg] : 0.0f;
#pragma unroll
      for (int e = 0; e < 8; ++e) {
        int rowg = row0 + wm + i * 16 + half * 8 + e;
        float val = acc[i][j][e] + bv;
        if (relu) val = val > 0.0f ? val : 0.0f;
        size_t idx = (size_t)rowg * N + colg;
        if (Cf) Cf[idx] = val;
        if (Cb) Cb[idx] = (bf16)val;
      }
    }
  }
}

// ---------------------------------------------------------------------------
// M[b,h,l] = max_s(q_l . K_sample_s) - sum_s(q_l . K_sample_s)/L
// ---------------------------------------------------------------------------
__global__ void probe_scores_kernel(const float* __restrict__ q,
                                    const float* __restrict__ k,
                                    const int* __restrict__ idxs,
                                    float* __restrict__ Mout)
{
  int gid = blockIdx.x * blockDim.x + threadIdx.x;
  if (gid >= B_ * H_ * L_) return;
  int l = gid % L_;
  int h = (gid / L_) % H_;
  int b = gid / (H_ * L_);
  const float* qr = q + (size_t)(b * L_ + l) * D_ + h * DH_;
  float qv[DH_];
#pragma unroll
  for (int d = 0; d < DH_; ++d) qv[d] = qr[d];
  float mx = -FLT_MAX, sm = 0.0f;
  int jnext = idxs[l * SK_];
  for (int s = 0; s < SK_; ++s) {
    int j = jnext;
    if (s + 1 < SK_) {
      jnext = idxs[l * SK_ + s + 1];
      __builtin_prefetch(k + (size_t)(b * L_ + jnext) * D_ + h * DH_, 0, 3);
    }
    const float* kr = k + (size_t)(b * L_ + j) * D_ + h * DH_;
    float dot = 0.0f;
#pragma unroll
    for (int d = 0; d < DH_; ++d) dot += qv[d] * kr[d];
    mx = fmaxf(mx, dot);
    sm += dot;
  }
  Mout[gid] = mx - sm * (1.0f / L_);
}

// ---------------------------------------------------------------------------
// Iterative top-40 arg-max per (b,h); masks selected entries in Mbuf.
// ---------------------------------------------------------------------------
__global__ void __launch_bounds__(256) topk_kernel(float* __restrict__ Mbuf,
                                                   int* __restrict__ Mtop)
{
  __shared__ float sv[256];
  __shared__ int   si[256];
  int bh  = blockIdx.x;
  int tid = threadIdx.x;
  float* Mrow = Mbuf + (size_t)bh * L_;
  for (int t = 0; t < NT_; ++t) {
    float bestv = -FLT_MAX; int besti = 0;
    for (int l = tid; l < L_; l += 256) {
      float val = Mrow[l];
      if (val > bestv) { bestv = val; besti = l; }
    }
    sv[tid] = bestv; si[tid] = besti;
    __syncthreads();
    for (int s = 128; s > 0; s >>= 1) {
      if (tid < s) {
        float v2 = sv[tid + s]; int i2 = si[tid + s];
        if (v2 > sv[tid] || (v2 == sv[tid] && i2 < si[tid])) { sv[tid] = v2; si[tid] = i2; }
      }
      __syncthreads();
    }
    if (tid == 0) { Mtop[bh * NT_ + t] = si[0]; Mrow[si[0]] = -FLT_MAX; }
    __syncthreads();
  }
}

// ---------------------------------------------------------------------------
// vmean[b,h,d] = mean_l v[b,l,h,d]
// ---------------------------------------------------------------------------
__global__ void __launch_bounds__(256) vmean_kernel(const float* __restrict__ v,
                                                    float* __restrict__ vmean)
{
  __shared__ float acc[256];
  int bh = blockIdx.x; int b = bh / H_, h = bh % H_;
  int tid = threadIdx.x;
  int d = tid & 31, chunk = tid >> 5;
  float s = 0.0f;
  for (int l = chunk; l < L_; l += 8)
    s += v[(size_t)(b * L_ + l) * D_ + h * DH_ + d];
  acc[tid] = s;
  __syncthreads();
  if (tid < DH_) {
    float t = 0.0f;
#pragma unroll
    for (int c = 0; c < 8; ++c) t += acc[c * 32 + tid];
    vmean[bh * DH_ + tid] = t * (1.0f / L_);
  }
}

// ---------------------------------------------------------------------------
// Per selected query u: scores = qT K * scale, softmax over L, ctx = attn . V
// One 256-thread block per (b,h,u); LDS softmax + ds_add_f32 accumulation.
// ---------------------------------------------------------------------------
__global__ void __launch_bounds__(256) topattn_kernel(
    const float* __restrict__ q, const float* __restrict__ k,
    const float* __restrict__ v, const int* __restrict__ Mtop,
    float* __restrict__ ctx_top)
{
  __shared__ float sc[L_];
  __shared__ float red[256];
  __shared__ float qs[DH_];
  __shared__ float ctx[DH_];
  int blk = blockIdx.x;           // bh*NT_ + u
  int bh  = blk / NT_;
  int b = bh / H_, h = bh % H_;
  int tid = threadIdx.x;
  int row = Mtop[blk];
  if (tid < DH_) {
    qs[tid]  = q[(size_t)(b * L_ + row) * D_ + h * DH_ + tid];
    ctx[tid] = 0.0f;
  }
  __syncthreads();
  const float scale = rsqrtf((float)DH_);
  float lmax = -FLT_MAX;
  for (int l = tid; l < L_; l += 256) {
    const float* kr = k + (size_t)(b * L_ + l) * D_ + h * DH_;
    float dot = 0.0f;
#pragma unroll
    for (int d = 0; d < DH_; ++d) dot += qs[d] * kr[d];
    dot *= scale;
    sc[l] = dot;
    lmax = fmaxf(lmax, dot);
  }
  red[tid] = lmax; __syncthreads();
  for (int s = 128; s > 0; s >>= 1) {
    if (tid < s) red[tid] = fmaxf(red[tid], red[tid + s]);
    __syncthreads();
  }
  float gmax = red[0]; __syncthreads();
  float lsum = 0.0f;
  for (int l = tid; l < L_; l += 256) {
    float e = __expf(sc[l] - gmax);
    sc[l] = e;
    lsum += e;
  }
  red[tid] = lsum; __syncthreads();
  for (int s = 128; s > 0; s >>= 1) {
    if (tid < s) red[tid] += red[tid + s];
    __syncthreads();
  }
  float inv = 1.0f / red[0];
  float part[DH_];
#pragma unroll
  for (int d = 0; d < DH_; ++d) part[d] = 0.0f;
  for (int l = tid; l < L_; l += 256) {
    float w = sc[l];
    const float* vr = v + (size_t)(b * L_ + l) * D_ + h * DH_;
#pragma unroll
    for (int d = 0; d < DH_; ++d) part[d] += w * vr[d];
  }
#pragma unroll
  for (int d = 0; d < DH_; ++d) atomicAdd(&ctx[d], part[d]);
  __syncthreads();
  if (tid < DH_) ctx_top[(size_t)blk * DH_ + tid] = ctx[tid] * inv;
}

// ---------------------------------------------------------------------------
// context = broadcast mean(v); then scatter ctx_top rows.  bf16 output so it
// feeds the Wo WMMA GEMM directly.
// ---------------------------------------------------------------------------
__global__ void fill_context_kernel(const float* __restrict__ vmean,
                                    bf16* __restrict__ ctxb)
{
  int gid = blockIdx.x * blockDim.x + threadIdx.x;
  if (gid >= ROWS_ * D_) return;
  int col = gid % D_;
  int row = gid / D_;
  int b = row / L_;
  int h = col / DH_, d = col % DH_;
  ctxb[gid] = (bf16)vmean[(b * H_ + h) * DH_ + d];
}

__global__ void scatter_context_kernel(const float* __restrict__ ctx_top,
                                       const int* __restrict__ Mtop,
                                       bf16* __restrict__ ctxb)
{
  int gid = blockIdx.x * blockDim.x + threadIdx.x;
  if (gid >= B_ * H_ * NT_ * DH_) return;
  int d   = gid % DH_;
  int blk = gid / DH_;
  int bh  = blk / NT_;
  int b = bh / H_, h = bh % H_;
  int row = Mtop[blk];
  ctxb[(size_t)(b * L_ + row) * D_ + h * DH_ + d] = (bf16)ctx_top[gid];
}

// ---------------------------------------------------------------------------
// out = LayerNorm(a + b) * g + beta ; optional bf16 copy for next GEMM.
// One row (D_=256 elems) per 256-thread block.
// ---------------------------------------------------------------------------
__global__ void __launch_bounds__(256) add_ln_kernel(
    const float* __restrict__ a, const float* __restrict__ bres,
    const float* __restrict__ g, const float* __restrict__ beta,
    float* __restrict__ outf, bf16* __restrict__ outb)
{
  __shared__ float rs[256], rq[256];
  int row = blockIdx.x;
  int tid = threadIdx.x;
  size_t idx = (size_t)row * D_ + tid;
  float val = a[idx] + bres[idx];
  rs[tid] = val; rq[tid] = val * val;
  __syncthreads();
  for (int s = 128; s > 0; s >>= 1) {
    if (tid < s) { rs[tid] += rs[tid + s]; rq[tid] += rq[tid + s]; }
    __syncthreads();
  }
  float mean = rs[0] * (1.0f / D_);
  float var  = rq[0] * (1.0f / D_) - mean * mean;
  float o = (val - mean) * rsqrtf(var + EPS_) * g[tid] + beta[tid];
  if (outf) outf[idx] = o;
  if (outb) outb[idx] = (bf16)o;
}

// ---------------------------------------------------------------------------
// Host orchestration
// ---------------------------------------------------------------------------
extern "C" void kernel_launch(void* const* d_in, const int* in_sizes, int n_in,
                              void* d_out, int out_size, void* d_ws, size_t ws_size,
                              hipStream_t stream)
{
  (void)in_sizes; (void)n_in; (void)out_size; (void)ws_size;
  const float* x    = (const float*)d_in[0];
  const float* Wq   = (const float*)d_in[1];
  const float* bq   = (const float*)d_in[2];
  const float* Wk   = (const float*)d_in[3];
  const float* bk   = (const float*)d_in[4];
  const float* Wv   = (const float*)d_in[5];
  const float* bv   = (const float*)d_in[6];
  const float* Wo   = (const float*)d_in[7];
  const float* bo   = (const float*)d_in[8];
  const float* c1w  = (const float*)d_in[9];
  const float* c1b  = (const float*)d_in[10];
  const float* c2w  = (const float*)d_in[11];
  const float* c2b  = (const float*)d_in[12];
  const float* ln1g = (const float*)d_in[13];
  const float* ln1b = (const float*)d_in[14];
  const float* ln2g = (const float*)d_in[15];
  const float* ln2b = (const float*)d_in[16];
  const int*   idxs = (const int*)d_in[17];

  char* ws = (char*)d_ws;
  float* qbuf = (float*)(ws);                      // 8MB : q, later att
  float* kbuf = (float*)(ws + (8u  << 20));        // 8MB : k, later x1 (post-LN1)
  float* vbuf = (float*)(ws + (16u << 20));        // 8MB : v, later y2 (FFN out)
  bf16*  xb   = (bf16*) (ws + (24u << 20));        // 4MB : x_bf16 -> context_bf16 -> x1_bf16
  bf16*  y1b  = (bf16*) (ws + (28u << 20));        // 8MB : FFN hidden (8192x512 bf16)
  bf16*  wWq  = (bf16*) (ws + (36u << 20));        // N-major bf16 weights (1MB)
  bf16*  wWk  = wWq + 256 * 256;
  bf16*  wWv  = wWk + 256 * 256;
  bf16*  wWo  = wWv + 256 * 256;
  bf16*  wC1  = wWo + 256 * 256;                   // (N=512, K=256)
  bf16*  wC2  = wC1 + 512 * 256;                   // (N=256, K=512)
  float* Mbuf = (float*)(ws + (37u << 20));        // 256KB : probe scores
  int*   Mtop = (int*)  (ws + (37u << 20) + (256u << 10));  // 5KB (padded 8KB)
  float* ctxt = (float*)(ws + (37u << 20) + (264u << 10));  // 160KB : ctx_top
  float* vmn  = (float*)(ws + (37u << 20) + (424u << 10));  // 4KB  : mean(v)

  // --- precision converts (weights -> N-major bf16) ------------------------
  f32_to_bf16_kernel<<<(ROWS_ * D_) / 256, 256, 0, stream>>>(x, xb, ROWS_ * D_);
  transpose_to_bf16_kernel<<<(256 * 256) / 256, 256, 0, stream>>>(Wq, wWq, 256, 256);
  transpose_to_bf16_kernel<<<(256 * 256) / 256, 256, 0, stream>>>(Wk, wWk, 256, 256);
  transpose_to_bf16_kernel<<<(256 * 256) / 256, 256, 0, stream>>>(Wv, wWv, 256, 256);
  transpose_to_bf16_kernel<<<(256 * 256) / 256, 256, 0, stream>>>(Wo, wWo, 256, 256);
  f32_to_bf16_kernel<<<(512 * 256) / 256, 256, 0, stream>>>(c1w, wC1, 512 * 256); // already (N,K)
  f32_to_bf16_kernel<<<(256 * 512) / 256, 256, 0, stream>>>(c2w, wC2, 256 * 512); // already (N,K)

  // --- Q/K/V projections (WMMA, async-pipelined) ---------------------------
  dim3 g256(256 / 64, ROWS_ / 128);  // (4, 64)
  gemm_bf16_wmma_kernel<<<g256, 256, 0, stream>>>(xb, wWq, bq, qbuf, nullptr, ROWS_, 256, 256, 0);
  gemm_bf16_wmma_kernel<<<g256, 256, 0, stream>>>(xb, wWk, bk, kbuf, nullptr, ROWS_, 256, 256, 0);
  gemm_bf16_wmma_kernel<<<g256, 256, 0, stream>>>(xb, wWv, bv, vbuf, nullptr, ROWS_, 256, 256, 0);

  // --- ProbSparse selection + attention ------------------------------------
  probe_scores_kernel<<<(B_ * H_ * L_) / 256, 256, 0, stream>>>(qbuf, kbuf, idxs, Mbuf);
  topk_kernel<<<B_ * H_, 256, 0, stream>>>(Mbuf, Mtop);
  vmean_kernel<<<B_ * H_, 256, 0, stream>>>(vbuf, vmn);
  topattn_kernel<<<B_ * H_ * NT_, 256, 0, stream>>>(qbuf, kbuf, vbuf, Mtop, ctxt);
  fill_context_kernel<<<(ROWS_ * D_) / 256, 256, 0, stream>>>(vmn, xb);     // xb = context (bf16)
  scatter_context_kernel<<<(B_ * H_ * NT_ * DH_) / 256, 256, 0, stream>>>(ctxt, Mtop, xb);

  // --- output projection + residual + LN1 ----------------------------------
  gemm_bf16_wmma_kernel<<<g256, 256, 0, stream>>>(xb, wWo, bo, qbuf /*att*/, nullptr, ROWS_, 256, 256, 0);
  add_ln_kernel<<<ROWS_, 256, 0, stream>>>(x, qbuf, ln1g, ln1b, kbuf /*x1*/, xb /*x1 bf16*/);

  // --- FFN (WMMA x2, fused ReLU + bf16 hidden) + residual + LN2 ------------
  dim3 g512(512 / 64, ROWS_ / 128);  // (8, 64)
  gemm_bf16_wmma_kernel<<<g512, 256, 0, stream>>>(xb, wC1, c1b, nullptr, y1b, ROWS_, 512, 256, 1);
  gemm_bf16_wmma_kernel<<<g256, 256, 0, stream>>>(y1b, wC2, c2b, vbuf /*y2*/, nullptr, ROWS_, 256, 512, 0);
  add_ln_kernel<<<ROWS_, 256, 0, stream>>>(kbuf, vbuf, ln2g, ln2b, (float*)d_out, nullptr);
}
